// FP8Linear_43018392436736
// MI455X (gfx1250) — compile-verified, matrix operand order checked by reading, not run on hardware
//
#include <hip/hip_runtime.h>
#include <math.h>

// ---------------------------------------------------------------------------
// FP8 (e4m3) delayed-scaling Linear for MI455X (gfx1250, wave32, WMMA).
//   out[8192,4096] = deq(fp8(x*sx)) @ deq(fp8(w*sw))^T + bias
// fp8 WMMA GEMM (v_wmma_f32_16x16x128_fp8_fp8), f32 accumulate, double-
// buffered async global->LDS pipeline (ASYNCcnt).
// Workspace (~50.4 MB): [x_q: 32MB][w_q: 16MB][amax_red: 2 floats]
// ---------------------------------------------------------------------------

typedef __attribute__((ext_vector_type(16))) int   v16i;
typedef __attribute__((ext_vector_type(8)))  float v8f;
typedef int v4i_t __attribute__((vector_size(16)));   // matches builtin param type

#define FP8_MAX   448.0f
#define AMAX_EPS  1e-8f
#define MOMENTUM  0.95f

#define M_DIM 8192
#define N_DIM 4096
#define K_DIM 4096

#define BLK_M 128
#define BLK_N 64
#define BLK_K 128
#define LDSP  144               // LDS row pitch: 128 data + 16 pad (16B mult, bank-spread)
#define NK    (K_DIM / BLK_K)   // 32 K-steps

#if defined(__has_builtin)
#if __has_builtin(__builtin_amdgcn_global_load_async_to_lds_b128) && \
    __has_builtin(__builtin_amdgcn_s_wait_asynccnt)
#define HAVE_ASYNC_LDS 1
#endif
#endif

// ---------------- fp8 e4m3 conversion ----------------

__device__ __forceinline__ unsigned int e4m3_sw(float f) {
    unsigned u   = __float_as_uint(f);
    unsigned sgn = (u >> 24) & 0x80u;
    float a = fminf(fabsf(f), FP8_MAX);
    if (a < 0.015625f) {                       // below 2^-6: denormal range, step 2^-9
        int k = (int)rintf(a * 512.0f);        // 0..8 (8 rolls into E=1,m=0 naturally)
        return sgn | (unsigned)k;
    }
    int e; float m = frexpf(a, &e);            // a = m * 2^e, m in [0.5, 1)
    int q = (int)rintf(m * 16.0f);             // 8..16, RNE
    if (q == 16) { q = 8; e += 1; }
    int E = e + 6;
    if (E > 15) { E = 15; q = 14; }            // clamp to 448 (0x7E), avoid NaN 0x7F
    return sgn | (unsigned)((E << 3) | (q - 8));
}

__device__ __forceinline__ unsigned int pack4_e4m3(float a, float b, float c, float d) {
#if __has_builtin(__builtin_amdgcn_cvt_pk_fp8_f32)
    int w = 0;
    w = __builtin_amdgcn_cvt_pk_fp8_f32(a, b, w, false);  // bytes 0,1
    w = __builtin_amdgcn_cvt_pk_fp8_f32(c, d, w, true);   // bytes 2,3
    return (unsigned int)w;
#else
    return e4m3_sw(a) | (e4m3_sw(b) << 8) | (e4m3_sw(c) << 16) | (e4m3_sw(d) << 24);
#endif
}

// ---------------- quantize + amax reduction ----------------

__global__ void amax_init_kernel(float* red) {
    red[0] = 0.0f;
    red[1] = 0.0f;
}

__global__ __launch_bounds__(256)
void quantize_kernel(const float* __restrict__ src, unsigned char* __restrict__ dst,
                     const float* __restrict__ amax_old, float* __restrict__ amax_red,
                     int n4) {
    const float s = FP8_MAX / fmaxf(amax_old[0], AMAX_EPS);  // delayed scaling
    float lmax = 0.0f;
    const int stride = gridDim.x * blockDim.x;
    const float4* __restrict__ src4 = (const float4*)src;
    unsigned int* __restrict__ dst4 = (unsigned int*)dst;
    for (int i = blockIdx.x * blockDim.x + threadIdx.x; i < n4; i += stride) {
        float4 v = src4[i];
        lmax = fmaxf(lmax, fmaxf(fmaxf(fabsf(v.x), fabsf(v.y)),
                                 fmaxf(fabsf(v.z), fabsf(v.w))));
        float a = fminf(fmaxf(v.x * s, -FP8_MAX), FP8_MAX);
        float b = fminf(fmaxf(v.y * s, -FP8_MAX), FP8_MAX);
        float c = fminf(fmaxf(v.z * s, -FP8_MAX), FP8_MAX);
        float d = fminf(fmaxf(v.w * s, -FP8_MAX), FP8_MAX);
        dst4[i] = pack4_e4m3(a, b, c, d);
    }
    __shared__ float red[256];
    red[threadIdx.x] = lmax;
    __syncthreads();
    for (int off = 128; off > 0; off >>= 1) {
        if ((int)threadIdx.x < off)
            red[threadIdx.x] = fmaxf(red[threadIdx.x], red[threadIdx.x + off]);
        __syncthreads();
    }
    if (threadIdx.x == 0)   // |x| >= 0 so IEEE bits compare as signed ints
        atomicMax((int*)amax_red, __float_as_int(red[0]));
}

__global__ void amax_final_kernel(const float* __restrict__ old_in,
                                  const float* __restrict__ old_w,
                                  const float* __restrict__ red,
                                  float* __restrict__ tail) {
    int i = threadIdx.x;
    if (i < 2) {
        float r = red[i];
        if (!(r == r))   r = AMAX_EPS;   // nan -> eps
        if (r > 3.0e38f) r = FP8_MAX;    // +inf -> 448
        float o = (i == 0) ? old_in[0] : old_w[0];
        tail[i] = fmaxf(fmaxf(o * MOMENTUM, r), AMAX_EPS);
    }
}

// ---------------- fp8 WMMA GEMM ----------------
// grid = (N/64, M/128), block = 256 threads = 8 waves in a 4(M) x 2(N) grid.
// Each wave computes a 32x32 region = 2x2 tiles of 16x16.
// Double-buffered LDS: per-buffer A 128x144 (18KB) + B 64x144 (9KB) -> 54KB.

union Frag8 { v16i v; unsigned long long q[8]; };

__device__ __forceinline__ void copy16_to_lds(const unsigned char* g, unsigned char* l) {
#if defined(HAVE_ASYNC_LDS)
    __builtin_amdgcn_global_load_async_to_lds_b128(
        (__attribute__((address_space(1))) v4i_t*)g,
        (__attribute__((address_space(3))) v4i_t*)l, 0, 0);
#else
    *(uint4*)l = *(const uint4*)g;     // global_load_b128 + ds_store_b128
#endif
}

#define A_BYTES (BLK_M * LDSP)   // 18432
#define B_BYTES (BLK_N * LDSP)   //  9216

__global__ __launch_bounds__(256)
void fp8_gemm_kernel(const unsigned char* __restrict__ Aq,
                     const unsigned char* __restrict__ Bq,
                     const float* __restrict__ bias,
                     const float* __restrict__ amax_in,
                     const float* __restrict__ amax_w,
                     float* __restrict__ out) {
    __shared__ unsigned char lsA[2 * A_BYTES];
    __shared__ unsigned char lsB[2 * B_BYTES];

    const int tid   = threadIdx.x;
    const int lane  = tid & 31;
    const int wave  = tid >> 5;
    const int waveM = wave >> 1;        // 0..3  -> 32-row band
    const int waveN = wave & 1;         // 0..1  -> 32-col band
    const int half  = lane >> 4;        // K-chunk interleave half
    const int l16   = lane & 15;        // row-in-tile

    const long rowA0 = (long)blockIdx.y * BLK_M;  // M base
    const long rowB0 = (long)blockIdx.x * BLK_N;  // N base

    const float sx  = FP8_MAX / fmaxf(amax_in[0], AMAX_EPS);
    const float sw  = FP8_MAX / fmaxf(amax_w[0],  AMAX_EPS);
    const float inv = 1.0f / (sx * sw);

    v8f acc[2][2];
    #pragma unroll
    for (int mt = 0; mt < 2; ++mt)
        #pragma unroll
        for (int nt = 0; nt < 2; ++nt)
            acc[mt][nt] = (v8f)(0.0f);

    // cooperative copy geometry: 256 threads x 16B chunks
    const int ldRow = tid >> 3;         // 0..31
    const int ldCol = (tid & 7) << 4;   // 0..112 step 16

    // issue one K-tile's copies into buffer `buf` (6 async ops per thread/wave)
    auto issue_tile = [&](int k0, int buf) {
        unsigned char* dA = lsA + buf * A_BYTES;
        unsigned char* dB = lsB + buf * B_BYTES;
        #pragma unroll
        for (int i = 0; i < 4; ++i) {   // A: 128 rows
            const int r = ldRow + i * 32;
            copy16_to_lds(Aq + (rowA0 + r) * K_DIM + k0 + ldCol, dA + r * LDSP + ldCol);
        }
        #pragma unroll
        for (int i = 0; i < 2; ++i) {   // B: 64 rows
            const int r = ldRow + i * 32;
            copy16_to_lds(Bq + (rowB0 + r) * K_DIM + k0 + ldCol, dB + r * LDSP + ldCol);
        }
        if (k0 + BLK_K < K_DIM) {       // L2 prefetch of the tile we issue next iter
            __builtin_prefetch(Aq + (rowA0 + ldRow) * K_DIM + k0 + BLK_K + ldCol, 0, 1);
            __builtin_prefetch(Bq + (rowB0 + ldRow) * K_DIM + k0 + BLK_K + ldCol, 0, 1);
        }
    };

    issue_tile(0, 0);

    for (int kt = 0; kt < NK; ++kt) {
        const int cur = kt & 1;
        if (kt + 1 < NK) {
            issue_tile((kt + 1) * BLK_K, cur ^ 1);   // prefetch next tile into other buffer
#if defined(HAVE_ASYNC_LDS)
            __builtin_amdgcn_s_wait_asynccnt(6);     // in-order: current tile's 6 ops done
#endif
        } else {
#if defined(HAVE_ASYNC_LDS)
            __builtin_amdgcn_s_wait_asynccnt(0);
#endif
        }
        __syncthreads();   // all waves' LDS writes for `cur` visible

        const unsigned char* bufA = lsA + cur * A_BYTES;
        const unsigned char* bufB = lsB + cur * B_BYTES;

        // A fragments: lane holds row l16, 8-byte K-chunks at 16*c + 8*half
        Frag8 fa[2];
        #pragma unroll
        for (int mt = 0; mt < 2; ++mt) {
            const unsigned char* base = bufA + (waveM * 32 + mt * 16 + l16) * LDSP + half * 8;
            #pragma unroll
            for (int c = 0; c < 8; ++c)
                fa[mt].q[c] = *(const unsigned long long*)(base + c * 16);
        }
        #pragma unroll
        for (int nt = 0; nt < 2; ++nt) {
            Frag8 fb;   // B fragment: lane holds weight-row (B column), same chunking
            const unsigned char* base = bufB + (waveN * 32 + nt * 16 + l16) * LDSP + half * 8;
            #pragma unroll
            for (int c = 0; c < 8; ++c)
                fb.q[c] = *(const unsigned long long*)(base + c * 16);

            acc[0][nt] = __builtin_amdgcn_wmma_f32_16x16x128_fp8_fp8(
                fa[0].v, fb.v, (short)0, acc[0][nt], false, false);
            acc[1][nt] = __builtin_amdgcn_wmma_f32_16x16x128_fp8_fp8(
                fa[1].v, fb.v, (short)0, acc[1][nt], false, false);
        }
        __syncthreads();   // done reading `cur` before it is re-filled next iter
    }

    // epilogue: C 16x16 layout -> element (r + 8*half, l16) held in acc VGPR r
    #pragma unroll
    for (int mt = 0; mt < 2; ++mt) {
        const long row0 = rowA0 + waveM * 32 + mt * 16 + half * 8;
        #pragma unroll
        for (int nt = 0; nt < 2; ++nt) {
            const long col = rowB0 + waveN * 32 + nt * 16 + l16;
            const float bv = bias[col];
            #pragma unroll
            for (int r = 0; r < 8; ++r)
                out[(row0 + r) * N_DIM + col] = acc[mt][nt][r] * inv + bv;
        }
    }
}

// ---------------- launch ----------------

extern "C" void kernel_launch(void* const* d_in, const int* in_sizes, int n_in,
                              void* d_out, int out_size, void* d_ws, size_t ws_size,
                              hipStream_t stream) {
    const float* x       = (const float*)d_in[0];   // [8192, 4096]
    const float* w       = (const float*)d_in[1];   // [4096, 4096]
    const float* bias    = (const float*)d_in[2];   // [4096]
    const float* in_amax = (const float*)d_in[3];   // scalar
    const float* w_amax  = (const float*)d_in[4];   // scalar
    float* out = (float*)d_out;                     // [8192*4096 + 2]

    unsigned char* ws = (unsigned char*)d_ws;
    unsigned char* Aq = ws;                                          // 32 MB
    unsigned char* Bq = ws + (size_t)M_DIM * K_DIM;                  // 16 MB
    float* red = (float*)(ws + (size_t)M_DIM * K_DIM + (size_t)N_DIM * K_DIM);

    amax_init_kernel<<<1, 32, 0, stream>>>(red);

    quantize_kernel<<<2048, 256, 0, stream>>>(x, Aq, in_amax, red + 0,
                                              (M_DIM * K_DIM) / 4);
    quantize_kernel<<<1024, 256, 0, stream>>>(w, Bq, w_amax, red + 1,
                                              (N_DIM * K_DIM) / 4);

    dim3 grid(N_DIM / BLK_N, M_DIM / BLK_M);   // (64, 64)
    fp8_gemm_kernel<<<grid, 256, 0, stream>>>(Aq, Bq, bias, in_amax, w_amax, out);

    amax_final_kernel<<<1, 32, 0, stream>>>(in_amax, w_amax, red,
                                            out + (size_t)M_DIM * N_DIM);
}
